// Step3p5Attention_82145544503835
// MI455X (gfx1250) — matrix operand order repaired and checked by reading, not compiled, not run on hardware
//
#include <hip/hip_runtime.h>
#include <stdint.h>

typedef __bf16 bf16;
typedef bf16 v16bf __attribute__((ext_vector_type(16)));
typedef bf16 v8bf  __attribute__((ext_vector_type(8)));
typedef float v8f  __attribute__((ext_vector_type(8)));
typedef unsigned int u32x4 __attribute__((ext_vector_type(4)));
typedef int i32x4 __attribute__((ext_vector_type(4)));
typedef int i32x8 __attribute__((ext_vector_type(8)));

#define HIDDEN 4096
#define NH 32
#define NKV 8
#define HD 128
#define SEQ 2048
#define QKV_OUT ((NH + 2 * NKV) * HD) /* 6144 */

#if defined(__has_builtin)
#if __has_builtin(__builtin_amdgcn_tensor_load_to_lds) && \
    __has_builtin(__builtin_amdgcn_s_wait_tensorcnt)
#define USE_TDM 1
#endif
#endif

// ---------------------------------------------------------------------------
// GEMM: C[M,N] = A[M,K] x B[K,N], fp32 in/out, bf16 WMMA f32-accumulate.
// 128x128 block tile, K-step 32, 8 waves; wave owns 32x64 (2x4 WMMA tiles).
// Software-pipelined: next K-step's global tiles load into registers while
// the current step's 8 WMMAs execute.
// ---------------------------------------------------------------------------
#define BM 128
#define BN 128
#define BK 32

__global__ __launch_bounds__(256) void gemm_bf16(const float* __restrict__ A,
                                                 const float* __restrict__ B,
                                                 float* __restrict__ C,
                                                 int M, int N, int K) {
  __shared__ bf16 sA[BM][BK];   // A tile, row-major
  __shared__ bf16 sBt[BN][BK];  // B tile transposed: [n][k]

  const int tid = threadIdx.x;
  const int lane = tid & 31;
  const int wave = tid >> 5;
  const int wm = wave >> 1;  // 0..3 -> 32-row strip
  const int wn = wave & 1;   // 0..1 -> 64-col strip
  const int m0 = blockIdx.y * BM;
  const int n0 = blockIdx.x * BN;
  const int hi = lane >> 4;
  const int l15 = lane & 15;

  // Per-thread staging coordinates (float4 granularity, 32-bit offsets).
  const int ar = tid >> 3, ac4 = tid & 7;    // A: rows ar+32i, col4 ac4
  const int br = tid >> 5, bc4 = tid & 31;   // B: rows br+8i,  col4 bc4
  const float* Abase = A + (uint32_t)(m0 + ar) * (uint32_t)K + ac4 * 4;
  const float* Bbase = B + (uint32_t)br * (uint32_t)N + n0 + bc4 * 4;

  v8f zero = {0, 0, 0, 0, 0, 0, 0, 0};
  v8f acc[2][4];
#pragma unroll
  for (int i = 0; i < 2; ++i)
#pragma unroll
    for (int j = 0; j < 4; ++j) acc[i][j] = zero;

  // Prologue: stage first K tile into registers.
  float4 ra[4], rb[4];
#pragma unroll
  for (int i = 0; i < 4; ++i) {
    ra[i] = *(const float4*)(Abase + (uint32_t)(32 * i) * (uint32_t)K);
    rb[i] = *(const float4*)(Bbase + (uint32_t)(8 * i) * (uint32_t)N);
  }

  for (int k0 = 0; k0 < K; k0 += BK) {
    // Commit staged registers to LDS (f32 -> bf16 on the way in).
#pragma unroll
    for (int i = 0; i < 4; ++i) {
      sA[ar + 32 * i][ac4 * 4 + 0] = (bf16)ra[i].x;
      sA[ar + 32 * i][ac4 * 4 + 1] = (bf16)ra[i].y;
      sA[ar + 32 * i][ac4 * 4 + 2] = (bf16)ra[i].z;
      sA[ar + 32 * i][ac4 * 4 + 3] = (bf16)ra[i].w;
      sBt[bc4 * 4 + 0][br + 8 * i] = (bf16)rb[i].x;
      sBt[bc4 * 4 + 1][br + 8 * i] = (bf16)rb[i].y;
      sBt[bc4 * 4 + 2][br + 8 * i] = (bf16)rb[i].z;
      sBt[bc4 * 4 + 3][br + 8 * i] = (bf16)rb[i].w;
    }
    __syncthreads();

    // Kick off next tile's global loads (overlaps with WMMA below).
    if (k0 + BK < K) {
#pragma unroll
      for (int i = 0; i < 4; ++i) {
        ra[i] = *(const float4*)(Abase + (uint32_t)(k0 + BK) +
                                 (uint32_t)(32 * i) * (uint32_t)K);
        rb[i] = *(const float4*)(Bbase + (uint32_t)(k0 + BK + 8 * i) * (uint32_t)N);
      }
    }

    // A fragments (16x32: M = lane&15; K = j + 16*(lane>=16)).
    v16bf a[2];
#pragma unroll
    for (int mi = 0; mi < 2; ++mi) {
      const int m = wm * 32 + mi * 16 + l15;
      const v8bf lo = *(const v8bf*)&sA[m][hi * 16];
      const v8bf hp = *(const v8bf*)&sA[m][hi * 16 + 8];
#pragma unroll
      for (int j = 0; j < 8; ++j) { a[mi][j] = lo[j]; a[mi][8 + j] = hp[j]; }
    }
    // B fragments (32x16: N = lane&15; K = j + 16*(lane>=16)); one 32B read.
    v16bf b[4];
#pragma unroll
    for (int ni = 0; ni < 4; ++ni) {
      const int n = wn * 64 + ni * 16 + l15;
      b[ni] = *(const v16bf*)&sBt[n][hi * 16];
    }
#pragma unroll
    for (int mi = 0; mi < 2; ++mi)
#pragma unroll
      for (int ni = 0; ni < 4; ++ni)
        acc[mi][ni] = __builtin_amdgcn_wmma_f32_16x16x32_bf16(
            false, a[mi], false, b[ni], (short)0, acc[mi][ni], false, false);
    __syncthreads();
  }

  // Epilogue: C/D layout row = r + 8*(lane>=16), col = lane&15. 32-bit math.
  const uint32_t cbase =
      (uint32_t)(m0 + wm * 32 + 8 * hi) * (uint32_t)N + (uint32_t)(n0 + wn * 64 + l15);
#pragma unroll
  for (int mi = 0; mi < 2; ++mi)
#pragma unroll
    for (int ni = 0; ni < 4; ++ni)
#pragma unroll
      for (int r = 0; r < 8; ++r)
        C[cbase + (uint32_t)(mi * 16 + r) * (uint32_t)N + ni * 16] = acc[mi][ni][r];
}

// ---------------------------------------------------------------------------
// Gemma RMS-norm + NeoX RoPE; emit bf16 Q/K/V. grid=(S, 48), block=128.
// ---------------------------------------------------------------------------
__global__ __launch_bounds__(128) void normrope(
    const float* __restrict__ qkv, const int* __restrict__ positions,
    const float* __restrict__ qw, const float* __restrict__ kw,
    bf16* __restrict__ Qb, bf16* __restrict__ Kb, bf16* __restrict__ Vb) {
  const int s = blockIdx.x;
  const int h = blockIdx.y;  // [0,32)=Q, [32,40)=K, [40,48)=V
  const int t = threadIdx.x;
  const float x = qkv[(size_t)s * QKV_OUT + h * HD + t];

  if (h >= NH + NKV) {  // V head: straight bf16 convert
    Vb[(size_t)s * (NKV * HD) + (h - NH - NKV) * HD + t] = (bf16)x;
    return;
  }

  float sq = x * x;
#pragma unroll
  for (int off = 16; off; off >>= 1) sq += __shfl_xor(sq, off, 32);
  __shared__ float wred[4];
  if ((t & 31) == 0) wred[t >> 5] = sq;
  __syncthreads();
  const float mean = (wred[0] + wred[1] + wred[2] + wred[3]) * (1.0f / HD);
  const float scale = rsqrtf(mean + 1e-5f);
  const bool isq = (h < NH);
  const float w = isq ? qw[t] : kw[t];

  __shared__ float xs[HD];
  xs[t] = x * scale * (1.0f + w);
  __syncthreads();

  if (t < HD / 2) {
    const float pos = (float)positions[s];
    const float inv_freq = __powf(1.0e6f, -(float)t * (1.0f / 64.0f));
    float sn, c;
    __sincosf(pos * inv_freq, &sn, &c);
    const float x1 = xs[t], x2 = xs[t + 64];
    bf16* dst = isq ? (Qb + (size_t)s * (NH * HD) + h * HD)
                    : (Kb + (size_t)s * (NKV * HD) + (h - NH) * HD);
    dst[t] = (bf16)(x1 * c - x2 * sn);
    dst[t + 64] = (bf16)(x2 * c + x1 * sn);
  }
}

// ---------------------------------------------------------------------------
// gate = sigmoid(hidden @ Wg) : [S, NH]; one wave per token.
// ---------------------------------------------------------------------------
__global__ __launch_bounds__(32) void gate_k(const float* __restrict__ hidden,
                                             const float* __restrict__ Wg,
                                             float* __restrict__ gate) {
  const int s = blockIdx.x;
  const int h = threadIdx.x;
  const float* hr = hidden + (size_t)s * HIDDEN;
  float acc = 0.0f;
  for (int k = 0; k < HIDDEN; ++k) acc = fmaf(hr[k], Wg[(size_t)k * NH + h], acc);
  gate[(size_t)s * NH + h] = 1.0f / (1.0f + __expf(-acc));
}

// ---------------------------------------------------------------------------
// TDM: DMA a 32-row x 128-elem bf16 tile (row stride 1024 elems) into LDS,
// with LDS padding of 4 dwords every 64 dwords (row stride 272B) to rotate
// banks for the column gathers. D# per CDNA5 ISA.
// ---------------------------------------------------------------------------
#define VROW 136  /* 128 data + 8 pad bf16 per LDS row */

#ifdef USE_TDM
__device__ __forceinline__ void tdm_load_tile(unsigned lds_addr,
                                              const bf16* gptr) {
  const unsigned long long ga = (unsigned long long)(uintptr_t)gptr;
  u32x4 g0;
  g0[0] = 1u;                                   // count=1 (user descriptor)
  g0[1] = lds_addr;                             // LDS byte address
  g0[2] = (unsigned)(ga & 0xffffffffull);       // global_addr[31:0]
  g0[3] = (unsigned)((ga >> 32) & 0x01ffffffull) | (2u << 30);  // type=2
  i32x8 g1;
  // data_size=2B | pad_enable | pad_interval=64dw (code 5) | pad_amount=4dw (code 3)
  g1[0] = 0x00010000 | (1 << 20) | (5 << 22) | (3 << 25);
  g1[1] = 0x04000000;        // tensor_dim0 = 1024 (bits 79:48, low half here)
  g1[2] = 0x08000000;        // dim0 hi = 0; tensor_dim1 = 2048 (low half)
  g1[3] = 128 << 16;         // dim1 hi = 0; tile_dim0 = 128
  g1[4] = 32;                // tile_dim1 = 32; tile_dim2 = 0
  g1[5] = 1024;              // tensor_dim0_stride (48b) = 1024
  g1[6] = 0;
  g1[7] = 0;
  i32x4 z4 = {0, 0, 0, 0};
#if __clang_major__ >= 23
  i32x8 z8 = {0, 0, 0, 0, 0, 0, 0, 0};
  __builtin_amdgcn_tensor_load_to_lds(g0, g1, z4, z4, z8, 0);
#else
  __builtin_amdgcn_tensor_load_to_lds(g0, g1, z4, z4, 0);
#endif
}
#endif

// ---------------------------------------------------------------------------
// Flash attention (causal, GQA G=4) + sigmoid gating. One wave per
// (head, 16-query tile). Per 32-key tile: TDM-stage K/V into LDS (TENSORcnt),
// 8 WMMA for QK^T + 8 WMMA for PV; P transposed through LDS (s_wait_dscnt).
// ---------------------------------------------------------------------------
__global__ __launch_bounds__(32) void attn(const bf16* __restrict__ Qb,
                                           const bf16* __restrict__ Kb,
                                           const bf16* __restrict__ Vb,
                                           const float* __restrict__ gate,
                                           float* __restrict__ Og) {
  const int head = blockIdx.y;
  const int kv = head >> 2;  // GQA: kv head = head / G
  const int q0 = blockIdx.x * 16;
  const int lane = threadIdx.x;
  const int hi = lane >> 4;
  const int l15 = lane & 15;

  __shared__ bf16 sK[32][VROW];
  __shared__ bf16 sV[32][VROW];
  __shared__ bf16 Pl[16][32];

  // Q fragments: A-matrix 16x32 per head-dim chunk (4 chunks over HD=128).
  v16bf qf[4];
  {
    const bf16* qrow = Qb + (size_t)(q0 + l15) * (NH * HD) + head * HD;
#pragma unroll
    for (int c = 0; c < 4; ++c) {
      const v8bf lo = *(const v8bf*)(qrow + c * 32 + hi * 16);
      const v8bf hp = *(const v8bf*)(qrow + c * 32 + hi * 16 + 8);
#pragma unroll
      for (int j = 0; j < 8; ++j) { qf[c][j] = lo[j]; qf[c][8 + j] = hp[j]; }
    }
  }

  v8f zero = {0, 0, 0, 0, 0, 0, 0, 0};
  v8f o[8];
#pragma unroll
  for (int d = 0; d < 8; ++d) o[d] = zero;
  float mrow[8], lrow[8];
#pragma unroll
  for (int r = 0; r < 8; ++r) { mrow[r] = -1e30f; lrow[r] = 0.0f; }

  const float sc = 0.08838834764831845f;  // HD^-0.5
  const int nkt = (q0 >> 5) + 1;          // causal key-tile count

  for (int kt = 0; kt < nkt; ++kt) {
    const int kb = kt * 32;

    // Previous iteration's LDS reads must land before the DMA overwrites.
    __asm__ volatile("s_wait_dscnt 0x0" ::: "memory");
#ifdef USE_TDM
    tdm_load_tile((unsigned)(uintptr_t)&sK[0][0],
                  Kb + (size_t)kb * (NKV * HD) + kv * HD);
    tdm_load_tile((unsigned)(uintptr_t)&sV[0][0],
                  Vb + (size_t)kb * (NKV * HD) + kv * HD);
    __builtin_amdgcn_s_wait_tensorcnt(0);
#else
    {  // Fallback: each lane copies one 256B row of K and V.
      const bf16* gk = Kb + (size_t)(kb + lane) * (NKV * HD) + kv * HD;
      const bf16* gv = Vb + (size_t)(kb + lane) * (NKV * HD) + kv * HD;
#pragma unroll
      for (int i = 0; i < 8; ++i) {
        *(v16bf*)&sK[lane][i * 16] = *((const v16bf*)gk + i);
        *(v16bf*)&sV[lane][i * 16] = *((const v16bf*)gv + i);
      }
      __asm__ volatile("s_wait_dscnt 0x0" ::: "memory");
    }
#endif

    // QK^T: B-fragment = K^T (kdim-major), contiguous 32B LDS reads.
    v8f s0 = zero, s1 = zero;
#pragma unroll
    for (int c = 0; c < 4; ++c) {
      const v16bf b0 = *(const v16bf*)&sK[l15][c * 32 + hi * 16];
      const v16bf b1 = *(const v16bf*)&sK[16 + l15][c * 32 + hi * 16];
      s0 = __builtin_amdgcn_wmma_f32_16x16x32_bf16(false, qf[c], false, b0,
                                                   (short)0, s0, false, false);
      s1 = __builtin_amdgcn_wmma_f32_16x16x32_bf16(false, qf[c], false, b1,
                                                   (short)0, s1, false, false);
    }

    // Scale + causal mask + online softmax.
    float rmax[8];
#pragma unroll
    for (int r = 0; r < 8; ++r) {
      const int qrow = q0 + r + 8 * hi;
      float a0 = s0[r] * sc;
      float a1 = s1[r] * sc;
      if (kb + l15 > qrow) a0 = -1e30f;
      if (kb + 16 + l15 > qrow) a1 = -1e30f;
      s0[r] = a0;
      s1[r] = a1;
      rmax[r] = fmaxf(a0, a1);
    }
#pragma unroll
    for (int off = 1; off < 16; off <<= 1)
#pragma unroll
      for (int r = 0; r < 8; ++r)
        rmax[r] = fmaxf(rmax[r], __shfl_xor(rmax[r], off, 32));

    float alpha[8], lsum[8];
#pragma unroll
    for (int r = 0; r < 8; ++r) {
      const float mn = fmaxf(mrow[r], rmax[r]);
      alpha[r] = __expf(mrow[r] - mn);
      mrow[r] = mn;
      const float p0 = __expf(s0[r] - mn);
      const float p1 = __expf(s1[r] - mn);
      s0[r] = p0;
      s1[r] = p1;
      lsum[r] = p0 + p1;
    }
#pragma unroll
    for (int off = 1; off < 16; off <<= 1)
#pragma unroll
      for (int r = 0; r < 8; ++r) lsum[r] += __shfl_xor(lsum[r], off, 32);
#pragma unroll
    for (int r = 0; r < 8; ++r) lrow[r] = lrow[r] * alpha[r] + lsum[r];
#pragma unroll
    for (int d = 0; d < 8; ++d)
#pragma unroll
      for (int r = 0; r < 8; ++r) o[d][r] *= alpha[r];

    // P: D-layout -> LDS -> A-layout (single-wave WG: barrier is a NOP, so
    // enforce the cross-lane round-trip with the CDNA5 split DS counter).
#pragma unroll
    for (int r = 0; r < 8; ++r) {
      Pl[r + 8 * hi][l15] = (bf16)s0[r];
      Pl[r + 8 * hi][16 + l15] = (bf16)s1[r];
    }
    __asm__ volatile("s_wait_dscnt 0x0" ::: "memory");

    v16bf pf;
    {
      const v8bf lo = *(const v8bf*)&Pl[l15][hi * 16];
      const v8bf hp = *(const v8bf*)&Pl[l15][hi * 16 + 8];
#pragma unroll
      for (int j = 0; j < 8; ++j) { pf[j] = lo[j]; pf[8 + j] = hp[j]; }
    }

    // PV: V as B-fragment (key-major); gathers hit rotated banks (padding).
#pragma unroll
    for (int d = 0; d < 8; ++d) {
      v16bf vb;
#pragma unroll
      for (int j = 0; j < 16; ++j) vb[j] = sV[16 * hi + j][d * 16 + l15];
      o[d] = __builtin_amdgcn_wmma_f32_16x16x32_bf16(false, pf, false, vb,
                                                     (short)0, o[d], false, false);
    }
  }

  // Normalize, apply sigmoid gate, write f32 [S, NH*HD].
#pragma unroll
  for (int r = 0; r < 8; ++r) {
    const int row = q0 + r + 8 * hi;
    const float f = gate[(size_t)row * NH + head] / lrow[r];
#pragma unroll
    for (int d = 0; d < 8; ++d)
      Og[(size_t)row * (NH * HD) + head * HD + d * 16 + l15] = o[d][r] * f;
  }
}

// ---------------------------------------------------------------------------
extern "C" void kernel_launch(void* const* d_in, const int* in_sizes, int n_in,
                              void* d_out, int out_size, void* d_ws,
                              size_t ws_size, hipStream_t stream) {
  const float* hidden = (const float*)d_in[0];
  const int* positions = (const int*)d_in[1];
  const float* Wqkv = (const float*)d_in[2];
  const float* Wo = (const float*)d_in[3];
  const float* Wg = (const float*)d_in[4];
  const float* qw = (const float*)d_in[5];
  const float* kw = (const float*)d_in[6];
  float* out = (float*)d_out;

  // Workspace layout (~109 MB, 256B-aligned slabs).
  char* ws = (char*)d_ws;
  float* qkv = (float*)ws;                 ws += (size_t)SEQ * QKV_OUT * 4;
  bf16* Qb = (bf16*)ws;                    ws += (size_t)SEQ * NH * HD * 2;
  bf16* Kb = (bf16*)ws;                    ws += (size_t)SEQ * NKV * HD * 2;
  bf16* Vb = (bf16*)ws;                    ws += (size_t)SEQ * NKV * HD * 2;
  float* gate = (float*)ws;                ws += (size_t)SEQ * NH * 4;
  float* attnG = (float*)ws;               ws += (size_t)SEQ * NH * HD * 4;

  gemm_bf16<<<dim3(QKV_OUT / BN, SEQ / BM), 256, 0, stream>>>(
      hidden, Wqkv, qkv, SEQ, QKV_OUT, HIDDEN);
  normrope<<<dim3(SEQ, NH + 2 * NKV), 128, 0, stream>>>(qkv, positions, qw, kw,
                                                        Qb, Kb, Vb);
  gate_k<<<dim3(SEQ), 32, 0, stream>>>(hidden, Wg, gate);
  attn<<<dim3(SEQ / 16, NH), 32, 0, stream>>>(Qb, Kb, Vb, gate, attnG);
  gemm_bf16<<<dim3(HIDDEN / BN, SEQ / BM), 256, 0, stream>>>(attnG, Wo, out,
                                                             SEQ, HIDDEN, HIDDEN);
}
